// Spatial_Inter_SA_25031069401492
// MI455X (gfx1250) — matrix operand order repaired and checked by reading, not compiled, not run on hardware
//
#include <hip/hip_runtime.h>
#include <hip/hip_bf16.h>

// ---------- types / helpers ----------
typedef __attribute__((ext_vector_type(16))) __bf16 v16bf;
typedef __attribute__((ext_vector_type(8)))  float  v8f;

union V16U {
  uint4 q[2];
  v16bf v;
  unsigned short s[16];
};

__device__ __forceinline__ unsigned short f2bf(float f) {
  unsigned int u = __float_as_uint(f);
  u += 0x7FFFu + ((u >> 16) & 1u);           // round-to-nearest-even
  return (unsigned short)(u >> 16);
}

// load 16 bf16 halves from two 16-byte chunks
__device__ __forceinline__ v16bf ld2(const unsigned short* p0, const unsigned short* p1) {
  V16U u;
  u.q[0] = *(const uint4*)p0;
  u.q[1] = *(const uint4*)p1;
  return u.v;
}

__device__ __forceinline__ v8f wmma_bf16(v16bf a, v16bf b, v8f c) {
  return __builtin_amdgcn_wmma_f32_16x16x32_bf16(false, a, false, b, (short)0, c,
                                                 false, false);
}

// ---------- geometry ----------
// N=8 frames, C=64, H=W=256, DP=32, HEADS=8, HD=4, seq L=256, head-dim D=1024
#define HW 65536
#define SEQD 262144      // 256*1024 halves per (n,head) matrix
#define QKV_ONE 16777216 // halves per Q (or K or Vt): 64 problems * 256*1024
#define SST 257          // padded LDS stride for score stripe

// ====================================================================
// Kernel 0: f32 -> bf16 weight conversion (run once per launch)
// ====================================================================
__global__ __launch_bounds__(256) void kconv(const float* __restrict__ src,
                                             unsigned short* __restrict__ dst,
                                             int n) {
  int i = blockIdx.x * 256 + threadIdx.x;
  if (i < n) dst[i] = f2bf(src[i]);
}

// ====================================================================
// Kernel 1: LN1 + QKV projection via WMMA. 256 tokens / WG (16 M-tiles).
// Q,K stored [p][row][1024]; V stored TRANSPOSED [p][feat 1024][key 256]
// so k2b B-tiles are contiguous (no LDS transpose needed downstream).
// ====================================================================
__global__ __launch_bounds__(256) void k1_ln_qkv(
    const float* __restrict__ x, const float* __restrict__ g1,
    const float* __restrict__ b1, const unsigned short* __restrict__ wb,
    const float* __restrict__ bproj, unsigned short* __restrict__ qkv, int dir) {
  __shared__ __align__(16) unsigned short naS[256 * 40];  // LN'd tokens, bf16

  int tid = threadIdx.x;
  int T0 = blockIdx.x * 256;  // tokens T0..T0+255 share (n,hrow); wcol = local
  int n = T0 >> 16;
  int hrow = (T0 >> 8) & 255;

  // ---- LN phase: one thread per token ----
  {
    int wcol = tid;
    const float* xp = x + (size_t)n * 64 * HW + hrow * 256 + wcol;
    float xv[64];
    float m = 0.f;
#pragma unroll
    for (int c = 0; c < 64; ++c) { xv[c] = xp[(size_t)c * HW]; m += xv[c]; }
    m *= (1.f / 64.f);
    float v = 0.f;
#pragma unroll
    for (int c = 0; c < 64; ++c) { float d = xv[c] - m; v += d * d; }
    v *= (1.f / 64.f);
    float inv = rsqrtf(v + 1e-5f);
    int cb = dir ? 32 : 0;
#pragma unroll
    for (int k = 0; k < 32; ++k)
      naS[tid * 40 + k] =
          f2bf((xv[cb + k] - m) * inv * g1[cb + k] + b1[cb + k]);
  }
  __syncthreads();

  // ---- WMMA projection: M=256 (16 tiles), N=96 (6 tiles), K=32 ----
  int lane = tid & 31, wv = tid >> 5;
  int hi = lane >> 4, l15 = lane & 15;

#pragma unroll 1
  for (int i = 0; i < 2; ++i) {
    int mt = wv * 2 + i;
    const unsigned short* ap = &naS[(mt * 16 + l15) * 40 + (hi ? 8 : 0)];
    v16bf a = ld2(ap, ap + 16);
#pragma unroll 1
    for (int nt = 0; nt < 6; ++nt) {
      int o = nt * 16 + l15;
      const unsigned short* bp = wb + o * 32 + (hi ? 8 : 0);
      v16bf b = ld2(bp, bp + 16);
      v8f acc = {};
      acc = wmma_bf16(a, b, acc);
      float bias = bproj[o];
      int qsel = o >> 5;    // uniform per nt
      int head = (o >> 2) & 7;
      int e = o & 3;
      unsigned short* qb =
          qkv + (size_t)qsel * QKV_ONE + (size_t)(n * 8 + head) * SEQD;
#pragma unroll
      for (int gi = 0; gi < 8; ++gi) {
        int tl = mt * 16 + gi + hi * 8;  // == wcol
        int rowIdx = dir ? tl : hrow;    // sequence index (query/key)
        int colIdx = dir ? hrow : tl;    // cross-dim index
        size_t off = (qsel == 2)
                         ? ((size_t)(colIdx * 4 + e)) * 256 + rowIdx  // Vt
                         : (size_t)rowIdx * 1024 + colIdx * 4 + e;    // Q,K
        qb[off] = f2bf(acc[gi] + bias);
      }
    }
  }
}

// ====================================================================
// Kernel 2a: S = Q K^T / 32, softmax over 256 keys, write P (bf16).
// grid(64 problems, 8 query-stripes of 32 rows), 256 thr (8 waves).
// ====================================================================
__global__ __launch_bounds__(256) void k2a_scores(
    const unsigned short* __restrict__ Qb, const unsigned short* __restrict__ Kb,
    unsigned short* __restrict__ Pb) {
  __shared__ float S[32 * SST];
  __shared__ float red[32 * 8];
  __shared__ float rowmax[32];
  __shared__ float rowsum[32];

  int p = blockIdx.x;
  int M0g = blockIdx.y * 32;
  int tid = threadIdx.x;
  int lane = tid & 31, wv = tid >> 5;
  int hi = lane >> 4, l15 = lane & 15;

  const unsigned short* Q = Qb + (size_t)p * SEQD;
  const unsigned short* K = Kb + (size_t)p * SEQD;

  v8f acc[2][2] = {};
  const unsigned short* arow0 = Q + (size_t)(M0g + l15) * 1024 + (hi ? 8 : 0);
  const unsigned short* arow1 = arow0 + 16 * 1024;
  const unsigned short* brow0 = K + (size_t)(wv * 32 + l15) * 1024 + (hi ? 8 : 0);
  const unsigned short* brow1 = brow0 + 16 * 1024;

  for (int kk = 0; kk < 1024; kk += 32) {
    v16bf a0 = ld2(arow0 + kk, arow0 + kk + 16);
    v16bf a1 = ld2(arow1 + kk, arow1 + kk + 16);
    v16bf b0 = ld2(brow0 + kk, brow0 + kk + 16);
    v16bf b1 = ld2(brow1 + kk, brow1 + kk + 16);
    acc[0][0] = wmma_bf16(a0, b0, acc[0][0]);
    acc[0][1] = wmma_bf16(a0, b1, acc[0][1]);
    acc[1][0] = wmma_bf16(a1, b0, acc[1][0]);
    acc[1][1] = wmma_bf16(a1, b1, acc[1][1]);
  }
#pragma unroll
  for (int m = 0; m < 2; ++m)
#pragma unroll
    for (int nt = 0; nt < 2; ++nt) {
      int col = wv * 32 + nt * 16 + l15;
#pragma unroll
      for (int gi = 0; gi < 8; ++gi) {
        int row = m * 16 + gi + hi * 8;
        S[row * SST + col] = acc[m][nt][gi] * 0.03125f;  // 1/sqrt(1024)
      }
    }
  __syncthreads();

  // row softmax; row = tid&31 => 32 distinct LDS banks per wave (stride 257)
  int r = tid & 31, sgp = tid >> 5;
  float mx = -3.4e38f;
  for (int j = 0; j < 32; ++j) mx = fmaxf(mx, S[r * SST + sgp * 32 + j]);
  red[r * 8 + sgp] = mx;
  __syncthreads();
  if (tid < 32) {
    float m2 = red[tid * 8];
    for (int j = 1; j < 8; ++j) m2 = fmaxf(m2, red[tid * 8 + j]);
    rowmax[tid] = m2;
  }
  __syncthreads();
  float rm = rowmax[r];
  float sm = 0.f;
  for (int j = 0; j < 32; ++j) {
    float e = __expf(S[r * SST + sgp * 32 + j] - rm);
    S[r * SST + sgp * 32 + j] = e;
    sm += e;
  }
  red[r * 8 + sgp] = sm;
  __syncthreads();
  if (tid < 32) {
    float s2 = 0.f;
    for (int j = 0; j < 8; ++j) s2 += red[tid * 8 + j];
    rowsum[tid] = s2;
  }
  __syncthreads();

  // P write: remap so 8 consecutive threads cover one row => coalesced stores
  int r2 = tid >> 3, sg2 = tid & 7;
  float invs = 1.0f / rowsum[r2];
  unsigned short* Pr = Pb + (size_t)p * 65536 + (size_t)(M0g + r2) * 256 + sg2 * 32;
  for (int j = 0; j < 32; ++j) Pr[j] = f2bf(S[r2 * SST + sg2 * 32 + j] * invs);
}

// ====================================================================
// Kernel 2b: O = P x Vt^T (M=256, N=1024, K=256). Vt is [feat][key] so
// B-tiles are contiguous global loads: barrier-free pipelined loop.
// grid(64 problems, 4 M-blocks of 64, 8 N-blocks of 128), 256 thr.
// Each wave: 2 M-tiles x 2 N-tiles (4 loads -> 4 WMMAs per K-step).
// ====================================================================
__global__ __launch_bounds__(256) void k2b_pv(
    const unsigned short* __restrict__ Pb, const unsigned short* __restrict__ Vt,
    unsigned short* __restrict__ Ob) {
  int p = blockIdx.x;
  int M0 = blockIdx.y * 64;
  int NC0 = blockIdx.z * 128;
  int tid = threadIdx.x;
  int lane = tid & 31, wv = tid >> 5;
  int hi = lane >> 4, l15 = lane & 15;

  const unsigned short* P = Pb + (size_t)p * 65536;
  const unsigned short* V = Vt + (size_t)p * SEQD;

  int mtg = wv & 1;   // 2 M-tile group
  int ntg = wv >> 1;  // 4 N-tile groups (cols ntg*32)
  v8f acc[2][2] = {};

  const unsigned short* arow0 =
      P + (size_t)(M0 + mtg * 32 + l15) * 256 + (hi ? 8 : 0);
  const unsigned short* arow1 = arow0 + 16 * 256;
  const unsigned short* brow0 =
      V + (size_t)(NC0 + ntg * 32 + l15) * 256 + (hi ? 8 : 0);
  const unsigned short* brow1 = brow0 + 16 * 256;

  for (int kk = 0; kk < 256; kk += 32) {
    v16bf a0 = ld2(arow0 + kk, arow0 + kk + 16);
    v16bf a1 = ld2(arow1 + kk, arow1 + kk + 16);
    v16bf b0 = ld2(brow0 + kk, brow0 + kk + 16);
    v16bf b1 = ld2(brow1 + kk, brow1 + kk + 16);
    acc[0][0] = wmma_bf16(a0, b0, acc[0][0]);
    acc[0][1] = wmma_bf16(a0, b1, acc[0][1]);
    acc[1][0] = wmma_bf16(a1, b0, acc[1][0]);
    acc[1][1] = wmma_bf16(a1, b1, acc[1][1]);
  }

  unsigned short* O = Ob + (size_t)p * SEQD;
#pragma unroll
  for (int m = 0; m < 2; ++m)
#pragma unroll
    for (int nt = 0; nt < 2; ++nt) {
      int col = NC0 + ntg * 32 + nt * 16 + l15;
#pragma unroll
      for (int gi = 0; gi < 8; ++gi) {
        int row = M0 + mtg * 32 + m * 16 + gi + hi * 8;
        O[(size_t)row * 1024 + col] = f2bf(acc[m][nt][gi]);
      }
    }
}

// ====================================================================
// Kernel 3: fuse GEMM + residual + LN2 + FFN (fc1/GELU/fc2) + residual,
// output in (n,c,h,w). 64 tokens per workgroup, 256 threads.
// ====================================================================
__global__ __launch_bounds__(256) void k3_fuse_ffn(
    const float* __restrict__ x, const unsigned short* __restrict__ OH,
    const unsigned short* __restrict__ OV,
    const unsigned short* __restrict__ wb_fuse, const float* __restrict__ b_fuse,
    const float* __restrict__ ln2g, const float* __restrict__ ln2b,
    const unsigned short* __restrict__ wb_fc1, const float* __restrict__ b_fc1,
    const unsigned short* __restrict__ wb_fc2, const float* __restrict__ b_fc2,
    float* __restrict__ out) {
  __shared__ __align__(16) unsigned short sh_ff1[64 * 264];  // also aliases xc
  __shared__ __align__(16) unsigned short sh_yn[64 * 88];
  __shared__ __align__(16) float sh_y[64 * 68];
  __shared__ float red2[64 * 4];

  int tid = threadIdx.x;
  int lane = tid & 31, wv = tid >> 5;
  int hi = lane >> 4, l15 = lane & 15;

  int T0 = blockIdx.x * 64;
  int n = T0 >> 16, rem = T0 & 65535;
  int hrow = rem >> 8, wcol0 = rem & 255;

  unsigned short* xc = sh_ff1;  // [64][stride 88] bf16, dead before ff1 written

  // Phase 1: build concat(oh,ov) and residual xh
#pragma unroll 1
  for (int j = 0; j < 16; ++j) {
    int flat = tid + 256 * j;           // 4096 elems
    int tl = flat & 63, c = flat >> 6;  // token-local, channel
    unsigned short val;
    if (c < 32) {
      int head = c >> 2, e = c & 3;
      val = OH[((size_t)(n * 8 + head) * 256 + hrow) * 1024 +
               (size_t)(wcol0 + tl) * 4 + e];
    } else {
      int head = (c - 32) >> 2, e = c & 3;
      val = OV[((size_t)(n * 8 + head) * 256 + (wcol0 + tl)) * 1024 +
               (size_t)hrow * 4 + e];
    }
    xc[tl * 88 + c] = val;
  }
#pragma unroll 1
  for (int j = 0; j < 16; ++j) {
    int flat = tid + 256 * j;
    int c = flat >> 6, tl = flat & 63;  // coalesced in tl
    sh_y[tl * 68 + c] =
        x[((size_t)(n * 64 + c) * 256 + hrow) * 256 + wcol0 + tl];
  }
  __syncthreads();

  // Phase 2: fuse GEMM: y += xc @ w_fuse^T + b_fuse (M=64,N=64,K=64)
#pragma unroll 1
  for (int i = 0; i < 2; ++i) {
    int ti = wv * 2 + i, mt = ti >> 2, nt = ti & 3;
    int N0 = nt * 16;
    v8f acc = {};
#pragma unroll
    for (int kk = 0; kk < 64; kk += 32) {
      const unsigned short* ap = &xc[(mt * 16 + l15) * 88 + kk + (hi ? 8 : 0)];
      v16bf a = ld2(ap, ap + 16);
      const unsigned short* bp = wb_fuse + (N0 + l15) * 64 + kk + (hi ? 8 : 0);
      v16bf b = ld2(bp, bp + 16);
      acc = wmma_bf16(a, b, acc);
    }
    int c = N0 + l15;
    float bias = b_fuse[c];
#pragma unroll
    for (int gi = 0; gi < 8; ++gi) {
      int tr = mt * 16 + gi + hi * 8;
      sh_y[tr * 68 + c] += acc[gi] + bias;
    }
  }
  __syncthreads();

  // Phase 3: LN2 -> yn (bf16); 4 threads per token row
  {
    int tr = tid & 63, pp = tid >> 6;
    float s = 0.f;
    for (int c = pp * 16; c < pp * 16 + 16; ++c) s += sh_y[tr * 68 + c];
    red2[tr * 4 + pp] = s;
    __syncthreads();
    float m = (red2[tr * 4] + red2[tr * 4 + 1] + red2[tr * 4 + 2] +
               red2[tr * 4 + 3]) * (1.f / 64.f);
    __syncthreads();
    float s2 = 0.f;
    for (int c = pp * 16; c < pp * 16 + 16; ++c) {
      float d = sh_y[tr * 68 + c] - m;
      s2 += d * d;
    }
    red2[tr * 4 + pp] = s2;
    __syncthreads();
    float var = (red2[tr * 4] + red2[tr * 4 + 1] + red2[tr * 4 + 2] +
                 red2[tr * 4 + 3]) * (1.f / 64.f);
    float inv = rsqrtf(var + 1e-5f);
    for (int c = pp * 16; c < pp * 16 + 16; ++c)
      sh_yn[tr * 88 + c] =
          f2bf((sh_y[tr * 68 + c] - m) * inv * ln2g[c] + ln2b[c]);
  }
  __syncthreads();

  // Phase 4: fc1 + GELU: ff1 = gelu(yn @ w_fc1^T + b_fc1)  (M=64,N=256,K=64)
  {
    int mt = wv >> 1;
#pragma unroll 1
    for (int q = 0; q < 8; ++q) {
      int nt = (wv & 1) * 8 + q;
      int N0 = nt * 16;
      v8f acc = {};
#pragma unroll
      for (int kk = 0; kk < 64; kk += 32) {
        const unsigned short* ap = &sh_yn[(mt * 16 + l15) * 88 + kk + (hi ? 8 : 0)];
        v16bf a = ld2(ap, ap + 16);
        const unsigned short* bp = wb_fc1 + (N0 + l15) * 64 + kk + (hi ? 8 : 0);
        v16bf b = ld2(bp, bp + 16);
        acc = wmma_bf16(a, b, acc);
      }
      int o = N0 + l15;
      float bias = b_fc1[o];
#pragma unroll
      for (int gi = 0; gi < 8; ++gi) {
        int tr = mt * 16 + gi + hi * 8;
        float u = acc[gi] + bias;
        float gl = 0.5f * u * (1.0f + erff(u * 0.70710678f));
        sh_ff1[tr * 264 + o] = f2bf(gl);
      }
    }
  }
  __syncthreads();

  // Phase 5: fc2 + residual: y = y + ff1 @ w_fc2^T + b_fc2 (M=64,N=64,K=256)
#pragma unroll 1
  for (int i = 0; i < 2; ++i) {
    int ti = wv * 2 + i, mt = ti >> 2, nt = ti & 3;
    int N0 = nt * 16;
    v8f acc = {};
#pragma unroll 1
    for (int kk = 0; kk < 256; kk += 32) {
      const unsigned short* ap = &sh_ff1[(mt * 16 + l15) * 264 + kk + (hi ? 8 : 0)];
      v16bf a = ld2(ap, ap + 16);
      const unsigned short* bp = wb_fc2 + (N0 + l15) * 256 + kk + (hi ? 8 : 0);
      v16bf b = ld2(bp, bp + 16);
      acc = wmma_bf16(a, b, acc);
    }
    int c = N0 + l15;
    float bias = b_fc2[c];
#pragma unroll
    for (int gi = 0; gi < 8; ++gi) {
      int tr = mt * 16 + gi + hi * 8;
      sh_y[tr * 68 + c] = sh_y[tr * 68 + c] + acc[gi] + bias;
    }
  }
  __syncthreads();

  // Phase 6: coalesced store to (n,c,h,w)
#pragma unroll 1
  for (int j = 0; j < 16; ++j) {
    int flat = tid + 256 * j;
    int c = flat >> 6, tl = flat & 63;
    out[((size_t)(n * 64 + c) * 256 + hrow) * 256 + wcol0 + tl] =
        sh_y[tl * 68 + c];
  }
}

// ====================================================================
extern "C" void kernel_launch(void* const* d_in, const int* in_sizes, int n_in,
                              void* d_out, int out_size, void* d_ws,
                              size_t ws_size, hipStream_t stream) {
  const float* x      = (const float*)d_in[0];
  const float* ln1g   = (const float*)d_in[1];
  const float* ln1b   = (const float*)d_in[2];
  const float* w_ch   = (const float*)d_in[3];
  const float* b_ch   = (const float*)d_in[4];
  const float* w_cv   = (const float*)d_in[5];
  const float* b_cv   = (const float*)d_in[6];
  const float* w_fuse = (const float*)d_in[7];
  const float* b_fuse = (const float*)d_in[8];
  const float* ln2g   = (const float*)d_in[9];
  const float* ln2b   = (const float*)d_in[10];
  const float* w_fc1  = (const float*)d_in[11];
  const float* b_fc1  = (const float*)d_in[12];
  const float* w_fc2  = (const float*)d_in[13];
  const float* b_fc2  = (const float*)d_in[14];
  float* out = (float*)d_out;

  char* ws = (char*)d_ws;
  unsigned short* qkv = (unsigned short*)ws;                  // 96 MB (Q,K,Vt bf16)
  unsigned short* OH  = (unsigned short*)(ws + 100663296);    // 32 MB
  unsigned short* OV  = (unsigned short*)(ws + 134217728);    // 32 MB
  unsigned short* P   = (unsigned short*)(ws + 167772160);    // 8 MB
  unsigned short* wbf = (unsigned short*)(ws + 176160768);    // bf16 weights
  unsigned short* wb_ch   = wbf;           // 3072
  unsigned short* wb_cv   = wbf + 3072;    // 3072
  unsigned short* wb_fuse = wbf + 6144;    // 4096
  unsigned short* wb_fc1  = wbf + 10240;   // 16384
  unsigned short* wb_fc2  = wbf + 26624;   // 16384

  // one-time weight conversion to bf16
  kconv<<<12, 256, 0, stream>>>(w_ch, wb_ch, 3072);
  kconv<<<12, 256, 0, stream>>>(w_cv, wb_cv, 3072);
  kconv<<<16, 256, 0, stream>>>(w_fuse, wb_fuse, 4096);
  kconv<<<64, 256, 0, stream>>>(w_fc1, wb_fc1, 16384);
  kconv<<<64, 256, 0, stream>>>(w_fc2, wb_fc2, 16384);

  for (int dir = 0; dir < 2; ++dir) {
    k1_ln_qkv<<<2048, 256, 0, stream>>>(x, ln1g, ln1b, dir ? wb_cv : wb_ch,
                                        dir ? b_cv : b_ch, qkv, dir);
    k2a_scores<<<dim3(64, 8, 1), 256, 0, stream>>>(qkv, qkv + QKV_ONE, P);
    k2b_pv<<<dim3(64, 4, 8), 256, 0, stream>>>(P, qkv + 2 * QKV_ONE,
                                               dir ? OV : OH);
  }
  k3_fuse_ffn<<<8192, 256, 0, stream>>>(x, OH, OV, wb_fuse, b_fuse, ln2g, ln2b,
                                        wb_fc1, b_fc1, wb_fc2, b_fc2, out);
}
